// QuanvolutionClassifier_65481071400135
// MI455X (gfx1250) — matrix operand order repaired and checked by reading, not compiled, not run on hardware
//
#include <hip/hip_runtime.h>
#include <hip/hip_bf16.h>
#include <math.h>

typedef __attribute__((ext_vector_type(16))) _Float16 v16h;
typedef __attribute__((ext_vector_type(8)))  _Float16 v8h;
typedef __attribute__((ext_vector_type(4)))  _Float16 v4h;
typedef __attribute__((ext_vector_type(8)))  float    v8f;

#define N_B    2048
#define N_P    196          // 784/4 patches per image
#define KDIM   784
#define KPAD   800          // pad K to multiple of 32
#define KTILES 25           // 800 / 32
#define WS_WOFF 0           // swizzled W halves (25*512 = 12800 halves)
#define WS_FOFF 32768       // feats f16 [2048][800]

// ---------------------------------------------------------------------------
// Prep: (a) swizzle W[10,784] f32 -> f16 in the per-lane B order the head kernel uses
//           (32x16 f16 B: lane L holds col N=L&15, K = (L>>4)*16 + i, i=0..15)
//       (b) zero the K-pad columns 784..799 of featsH
// ---------------------------------------------------------------------------
__global__ void quanv_prep(const float* __restrict__ W,
                           _Float16* __restrict__ whSwz,
                           _Float16* __restrict__ featsH) {
    int t = blockIdx.x * blockDim.x + threadIdx.x;
    if (t < KTILES * 512) {
        int kt   = t >> 9;
        int r    = t & 511;
        int lane = r >> 4;
        int i    = r & 15;
        int n    = lane & 15;
        int k    = kt * 32 + ((lane >> 4) << 4) + i;
        float v  = (n < 10 && k < KDIM) ? W[n * KDIM + k] : 0.0f;
        whSwz[t] = (_Float16)v;
    }
    int u = t - KTILES * 512;
    if (u >= 0 && u < N_B * 16) {
        int row = u >> 4;
        int col = KDIM + (u & 15);
        featsH[(size_t)row * KPAD + col] = (_Float16)0.0f;
    }
}

// ---------------------------------------------------------------------------
// In-register 4-qubit gates. Index: i = q0*8 + q1*4 + q2*2 + q3 (wire0 = bit 8)
// ---------------------------------------------------------------------------
template<int M>
__device__ __forceinline__ void apply_ry(float* re, float* im, float c, float s) {
#pragma unroll
    for (int i = 0; i < 16; ++i) {
        if (i & M) continue;
        int j = i | M;
        float r0 = re[i], i0 = im[i], r1 = re[j], i1 = im[j];
        re[i] = c * r0 - s * r1;  im[i] = c * i0 - s * i1;
        re[j] = s * r0 + c * r1;  im[j] = s * i0 + c * i1;
    }
}
template<int M>
__device__ __forceinline__ void apply_rx(float* re, float* im, float c, float s) {
#pragma unroll
    for (int i = 0; i < 16; ++i) {
        if (i & M) continue;
        int j = i | M;
        float r0 = re[i], i0 = im[i], r1 = re[j], i1 = im[j];
        // a0' = c*a0 - i*s*a1 ; a1' = -i*s*a0 + c*a1
        re[i] = c * r0 + s * i1;  im[i] = c * i0 - s * r1;
        re[j] = c * r1 + s * i0;  im[j] = c * i1 - s * r0;
    }
}
template<int M>
__device__ __forceinline__ void apply_rz(float* re, float* im, float c, float s) {
#pragma unroll
    for (int i = 0; i < 16; ++i) {
        float r = re[i], q = im[i];
        if (i & M) { re[i] = c * r - s * q;  im[i] = c * q + s * r; }   // e^{+i t/2}
        else       { re[i] = c * r + s * q;  im[i] = c * q - s * r; }   // e^{-i t/2}
    }
}
template<int CM, int TM>
__device__ __forceinline__ void apply_cnot(float* re, float* im) {
#pragma unroll
    for (int i = 0; i < 16; ++i) {
        if ((i & CM) && !(i & TM)) {
            int j = i | TM;
            float tr = re[i]; re[i] = re[j]; re[j] = tr;
            float ti = im[i]; im[i] = im[j]; im[j] = ti;
        }
    }
}

// ---------------------------------------------------------------------------
// Fused conv(2x2,s2) + 4-qubit circuit + PauliZ measurement. One thread/patch.
// ---------------------------------------------------------------------------
__global__ void __launch_bounds__(256)
quanv_features_kernel(const float* __restrict__ x,
                      const float* __restrict__ cw,
                      const float* __restrict__ theta,
                      _Float16* __restrict__ featsH) {
    int g = blockIdx.x * blockDim.x + threadIdx.x;
    if (g >= N_B * N_P) return;
    int b  = g / N_P;
    int p  = g - b * N_P;
    int r  = p / 7;                 // conv-output row (0..27); 4 cols per patch
    int c0 = (p - r * 7) * 4;       // conv-output col base

    const float* xb = x + (size_t)b * 3136 + (2 * r) * 56 + 2 * c0;
    float4 r0a = *(const float4*)(xb);
    float4 r0b = *(const float4*)(xb + 4);
    float4 r1a = *(const float4*)(xb + 56);
    float4 r1b = *(const float4*)(xb + 60);
    float w00 = cw[0], w01 = cw[1], w10 = cw[2], w11 = cw[3];

    float a0 = r0a.x * w00 + r0a.y * w01 + r1a.x * w10 + r1a.y * w11;
    float a1 = r0a.z * w00 + r0a.w * w01 + r1a.z * w10 + r1a.w * w11;
    float a2 = r0b.x * w00 + r0b.y * w01 + r1b.x * w10 + r1b.y * w11;
    float a3 = r0b.z * w00 + r0b.w * w01 + r1b.z * w10 + r1b.w * w11;

    // Encoder RY(a_w) on wire w from |0000>: product state of (cos, sin)
    float ec[4], es[4];
    __sincosf(a0 * 0.5f, &es[0], &ec[0]);
    __sincosf(a1 * 0.5f, &es[1], &ec[1]);
    __sincosf(a2 * 0.5f, &es[2], &ec[2]);
    __sincosf(a3 * 0.5f, &es[3], &ec[3]);

    float re[16], im[16];
#pragma unroll
    for (int i = 0; i < 16; ++i) {
        float v = ((i & 8) ? es[0] : ec[0]) * ((i & 4) ? es[1] : ec[1]) *
                  ((i & 2) ? es[2] : ec[2]) * ((i & 1) ? es[3] : ec[3]);
        re[i] = v; im[i] = 0.0f;
    }

    float tc, ts;
    __sincosf(theta[0] * 0.5f, &ts, &tc); apply_ry<8>(re, im, tc, ts);   // RY w0
    __sincosf(theta[1] * 0.5f, &ts, &tc); apply_rx<4>(re, im, tc, ts);   // RX w1
    __sincosf(theta[2] * 0.5f, &ts, &tc); apply_rz<2>(re, im, tc, ts);   // RZ w2
    apply_cnot<8, 4>(re, im);                                            // CNOT 0->1
    __sincosf(theta[3] * 0.5f, &ts, &tc); apply_ry<1>(re, im, tc, ts);   // RY w3
    apply_cnot<2, 1>(re, im);                                            // CNOT 2->3
    __sincosf(theta[4] * 0.5f, &ts, &tc); apply_rx<8>(re, im, tc, ts);   // RX w0
    __sincosf(theta[5] * 0.5f, &ts, &tc); apply_rz<1>(re, im, tc, ts);   // RZ w3

    float z0 = 0.f, z1 = 0.f, z2 = 0.f, z3 = 0.f;
#pragma unroll
    for (int i = 0; i < 16; ++i) {
        float pr = re[i] * re[i] + im[i] * im[i];
        z0 += (i & 8) ? -pr : pr;
        z1 += (i & 4) ? -pr : pr;
        z2 += (i & 2) ? -pr : pr;
        z3 += (i & 1) ? -pr : pr;
    }
    v4h o = { (_Float16)z0, (_Float16)z1, (_Float16)z2, (_Float16)z3 };
    *(v4h*)(featsH + (size_t)b * KPAD + 4 * p) = o;
}

// ---------------------------------------------------------------------------
// Head: feats[16 rows, 800] x Wt[800, 16(pad)] via v_wmma_f32_16x16x32_f16,
// then bias + log_softmax. One wave (32 threads) per 16-row batch tile.
// ---------------------------------------------------------------------------
__global__ void __launch_bounds__(32)
quanv_head_kernel(const _Float16* __restrict__ featsH,
                  const _Float16* __restrict__ whSwz,
                  const float* __restrict__ bias,
                  float* __restrict__ out) {
    int tile = blockIdx.x;        // 0..127
    int lane = threadIdx.x;       // 0..31
    int n    = lane & 15;         // output column (class), pad >= 10
    int hi   = lane >> 4;         // half-wave id
    int rowA = tile * 16 + n;     // A-matrix row M = lane&15
    int kb   = hi * 16;           // A: lanes<16 hold K {0..15}; else {16..31}

    v8f acc = {};
    for (int kt = 0; kt < KTILES; ++kt) {
        const _Float16* ap = featsH + (size_t)rowA * KPAD + kt * 32 + kb;
        v8h alo = *(const v8h*)(ap);
        v8h ahi = *(const v8h*)(ap + 8);
        v16h a;
#pragma unroll
        for (int i = 0; i < 8; ++i) { a[i] = alo[i]; a[8 + i] = ahi[i]; }

        const _Float16* bp = whSwz + kt * 512 + lane * 16;
        v8h blo = *(const v8h*)(bp);
        v8h bhi = *(const v8h*)(bp + 8);
        v16h bm;
#pragma unroll
        for (int i = 0; i < 8; ++i) { bm[i] = blo[i]; bm[8 + i] = bhi[i]; }

        acc = __builtin_amdgcn_wmma_f32_16x16x32_f16(
            /*neg_a=*/false, a, /*neg_b=*/false, bm,
            /*c_mod=*/(short)0, acc, /*reuse_a=*/false, /*reuse_b=*/false);
    }

    float bv = (n < 10) ? bias[n] : 0.0f;
#pragma unroll
    for (int r = 0; r < 8; ++r) {
        // D layout: VGPR r holds row m = r + 8*hi, column n = lane&15
        float logit = acc[r] + bv;
        float mval = (n < 10) ? logit : -__builtin_inff();
#pragma unroll
        for (int msk = 1; msk < 16; msk <<= 1)
            mval = fmaxf(mval, __shfl_xor(mval, msk, 16));
        float e = (n < 10) ? __expf(logit - mval) : 0.0f;
#pragma unroll
        for (int msk = 1; msk < 16; msk <<= 1)
            e += __shfl_xor(e, msk, 16);
        if (n < 10) {
            int row = tile * 16 + hi * 8 + r;
            out[row * 10 + n] = logit - mval - __logf(e);
        }
    }
}

// ---------------------------------------------------------------------------
extern "C" void kernel_launch(void* const* d_in, const int* in_sizes, int n_in,
                              void* d_out, int out_size, void* d_ws, size_t ws_size,
                              hipStream_t stream) {
    const float* x     = (const float*)d_in[0];   // [2048,1,56,56]
    const float* cw    = (const float*)d_in[1];   // [1,1,2,2]
    const float* theta = (const float*)d_in[2];   // [6]
    const float* W     = (const float*)d_in[3];   // [10,784]
    const float* bias  = (const float*)d_in[4];   // [10]
    float* out = (float*)d_out;                   // [2048,10]

    _Float16* whSwz  = (_Float16*)((char*)d_ws + WS_WOFF);
    _Float16* featsH = (_Float16*)((char*)d_ws + WS_FOFF);

    int prepThreads = KTILES * 512 + N_B * 16;
    quanv_prep<<<(prepThreads + 255) / 256, 256, 0, stream>>>(W, whSwz, featsH);

    int nPatch = N_B * N_P;
    quanv_features_kernel<<<(nPatch + 255) / 256, 256, 0, stream>>>(x, cw, theta, featsH);

    quanv_head_kernel<<<N_B / 16, 32, 0, stream>>>(featsH, whSwz, bias, out);
}
